// TransformerBlock_90993177133623
// MI455X (gfx1250) — matrix operand order repaired and checked by reading, not compiled, not run on hardware
//
#include <hip/hip_runtime.h>
#include <hip/hip_bf16.h>
#include <math.h>

// ---------------------------------------------------------------------------
// Problem constants (from reference)
// ---------------------------------------------------------------------------
#define BB   4
#define SS   2048
#define EE   1024
#define HH   16
#define DHD  64
#define FF   4096
#define MM   (BB * SS)            // 8192 rows
#define EPSLN 1e-5f
#define SCALE_ATTN 8.0f           // sqrt(DH) -- reference MULTIPLIES

typedef __bf16 bf16_t;
typedef __attribute__((ext_vector_type(16))) __bf16 v16bf;
typedef __attribute__((ext_vector_type(8)))  float  v8f;

// Async copy: 16B global -> LDS, no VGPR round-trip (ASYNCcnt-tracked).
//   vdst = per-lane LDS byte address, vaddr = per-lane 32-bit byte offset,
//   saddr = uniform 64-bit base (GVS mode).
#define ASYNC_CP_B128(ldsoff, goff, base)                                   \
    asm volatile("global_load_async_to_lds_b128 %0, %1, %2"                 \
                 :: "v"(ldsoff), "v"(goff), "s"(base) : "memory")

#if __has_builtin(__builtin_amdgcn_s_wait_asynccnt)
#define WAIT_ASYNC(n) __builtin_amdgcn_s_wait_asynccnt(n)
#else
#define WAIT_ASYNC(n) asm volatile("s_wait_asynccnt %0" ::"i"(n) : "memory")
#endif

// ---------------------------------------------------------------------------
// fp32 -> bf16 conversion (weights / activations), grid-stride
// ---------------------------------------------------------------------------
__global__ void cvt_f32_bf16(const float* __restrict__ in,
                             bf16_t* __restrict__ out, int n) {
    for (int i = blockIdx.x * blockDim.x + threadIdx.x; i < n;
         i += gridDim.x * blockDim.x)
        out[i] = (bf16_t)in[i];
}

// ---------------------------------------------------------------------------
// LayerNorm over E=1024, one row per block (256 threads), bf16 output
// ---------------------------------------------------------------------------
__global__ __launch_bounds__(256) void layernorm_bf16(
    const float* __restrict__ x, const float* __restrict__ w,
    const float* __restrict__ b, bf16_t* __restrict__ y) {
    __shared__ float ssum[256];
    __shared__ float ssq[256];
    const int row = blockIdx.x;
    const int tid = threadIdx.x;
    const float* xr = x + (size_t)row * EE;

    float v[4];
    float s = 0.f, q = 0.f;
#pragma unroll
    for (int j = 0; j < 4; ++j) {
        v[j] = xr[tid + j * 256];
        s += v[j];
        q += v[j] * v[j];
    }
    ssum[tid] = s;
    ssq[tid]  = q;
    __syncthreads();
    for (int st = 128; st > 0; st >>= 1) {
        if (tid < st) {
            ssum[tid] += ssum[tid + st];
            ssq[tid]  += ssq[tid + st];
        }
        __syncthreads();
    }
    const float mu   = ssum[0] * (1.0f / EE);
    const float var  = ssq[0] * (1.0f / EE) - mu * mu;
    const float rstd = rsqrtf(var + EPSLN);

    bf16_t* yr = y + (size_t)row * EE;
#pragma unroll
    for (int j = 0; j < 4; ++j) {
        const int c = tid + j * 256;
        yr[c] = (bf16_t)((v[j] - mu) * rstd * w[c] + b[c]);
    }
}

// ---------------------------------------------------------------------------
// bf16 WMMA GEMM:  C[M,N] = A[M,K] @ W[N,K]^T + bias  (+ epilogue)
//   EPI 0 : store f32                      (QKV projections)
//   EPI 1 : store f32 + residual R[M,N]    (Wo and W2 output projections)
//   EPI 2 : exact GELU -> store bf16       (FFN first layer)
//
// Block tile 128(M) x 256(N), 256 threads = 8 waves in a 2x4 grid; each wave
// owns a 64x64 patch = 4x4 tiles of v_wmma_f32_16x16x32_bf16 (16 WMMA per
// K-step of 32). Global->LDS staging uses global_load_async_to_lds_b128 into
// a double buffer (2 x 24 KB), synchronized with s_wait_asynccnt + barriers.
// The last K-iteration is peeled so the steady-state loop has straight-line
// control flow: issue(k+1) -> wait(6) -> barrier -> 16 WMMA -> barrier.
// ---------------------------------------------------------------------------
__device__ __forceinline__ float gelu_exact(float v) {
    return 0.5f * v * (1.0f + erff(v * 0.70710678118654752f));
}

template <int EPI>
__global__ __launch_bounds__(256, 1) void gemm_bf16_wmma(
    const bf16_t* __restrict__ A,     // [M,K], K-contiguous
    const bf16_t* __restrict__ W,     // [N,K], K-contiguous (row n = B col n)
    const float*  __restrict__ bias,  // [N]
    const float*  __restrict__ R,     // [M,N] residual (EPI==1) or nullptr
    float*        __restrict__ Cf,    // f32 out (EPI 0/1)
    bf16_t*       __restrict__ Cb,    // bf16 out (EPI 2)
    int N, int K) {
    // LDS row = 32 bf16 (64 B) = 4 x uint4. Double-buffered A/B tiles.
    __shared__ uint4 sA[2][128 * 4];   // 2 x 8 KB
    __shared__ uint4 sB[2][256 * 4];   // 2 x 16 KB

    const int tid  = threadIdx.x;
    const int lane = tid & 31;
    const int wave = tid >> 5;
    const int wm   = wave & 1;   // 2 M-slices of 64 rows
    const int wn   = wave >> 1;  // 4 N-slices of 64 cols
    const int h    = lane >> 4;  // half-wave select
    const int l15  = lane & 15;

    const int bm = blockIdx.x;   // M / 128
    const int bn = blockIdx.y;   // N / 256

    // LDS byte addresses of the buffers (low 32 bits of the flat address are
    // the LDS offset per the aperture mapping).
    const unsigned ldsA0 = (unsigned)(unsigned long long)&sA[0][0];
    const unsigned ldsB0 = (unsigned)(unsigned long long)&sB[0][0];

    // Per-thread staging geometry (16B chunks; row = idx>>2, seg = idx&3).
    // A tile: 512 chunks (2/thread). B tile: 1024 chunks (4/thread).
    auto issue_tile = [&](int kb, int buf) {
#pragma unroll
        for (int i = 0; i < 2; ++i) {
            const int idx = tid + i * 256;
            const int r   = idx >> 2;
            const int seg = idx & 3;
            const unsigned goff =
                (unsigned)((bm * 128 + r) * K + kb * 32 + seg * 8) * 2u;
            const unsigned loff = ldsA0 + (unsigned)(buf * 512 + idx) * 16u;
            ASYNC_CP_B128(loff, goff, A);
        }
#pragma unroll
        for (int i = 0; i < 4; ++i) {
            const int idx = tid + i * 256;
            const int r   = idx >> 2;
            const int seg = idx & 3;
            const unsigned goff =
                (unsigned)((bn * 256 + r) * K + kb * 32 + seg * 8) * 2u;
            const unsigned loff = ldsB0 + (unsigned)(buf * 1024 + idx) * 16u;
            ASYNC_CP_B128(loff, goff, W);
        }
    };

    union Frag { uint4 u[2]; v16bf v; };

    const v8f vzero = {0.f, 0.f, 0.f, 0.f, 0.f, 0.f, 0.f, 0.f};
    v8f acc[4][4];
#pragma unroll
    for (int tm = 0; tm < 4; ++tm)
#pragma unroll
        for (int tn = 0; tn < 4; ++tn) acc[tm][tn] = vzero;

    // Fragment loads + 16 WMMAs for the tile in buffer `cur`.
    //   A (ISA 7.12.2, 16-bit 16x32): lanes 0-15 hold K0-7/K16-23,
    //   lanes 16-31 hold K8-15/K24-31. B (KxN from W rows): same K order,
    //   lanes 0-15 hold K0-7/K16-23 of column n, lanes 16-31 hold K8-15/K24-31.
    auto compute_tile = [&](int cur) {
        Frag af[4];
#pragma unroll
        for (int tm = 0; tm < 4; ++tm) {
            const int row = wm * 64 + tm * 16 + l15;
            af[tm].u[0] = sA[cur][row * 4 + h];
            af[tm].u[1] = sA[cur][row * 4 + h + 2];
        }
        Frag bf[4];
#pragma unroll
        for (int tn = 0; tn < 4; ++tn) {
            const int col = wn * 64 + tn * 16 + l15;
            bf[tn].u[0] = sB[cur][col * 4 + h];
            bf[tn].u[1] = sB[cur][col * 4 + h + 2];
        }
#pragma unroll
        for (int tm = 0; tm < 4; ++tm)
#pragma unroll
            for (int tn = 0; tn < 4; ++tn)
                acc[tm][tn] = __builtin_amdgcn_wmma_f32_16x16x32_bf16(
                    false, af[tm].v, false, bf[tn].v,
                    (short)0, acc[tm][tn], false, false);
    };

    const int ksteps = K >> 5;  // K / 32 (always >= 2 here)
    issue_tile(0, 0);           // prologue: 6 async loads in flight

    // Steady state: always prefetch next tile; no branches in the loop body.
    for (int kb = 0; kb < ksteps - 1; ++kb) {
        const int cur = kb & 1;
        issue_tile(kb + 1, 1 - cur);  // +6 in flight (total <= 12)
        WAIT_ASYNC(6);                // tile kb's 6 loads retired (in-order)
        __syncthreads();              // buffer `cur` visible to all 8 waves
        compute_tile(cur);
        __syncthreads();              // reads drained before buffer overwrite
    }

    // Peeled final iteration: nothing left to prefetch.
    WAIT_ASYNC(0);
    __syncthreads();
    compute_tile((ksteps - 1) & 1);

    // Epilogue. C/D layout: VGPR r -> M = r + 8*h ; lane&15 -> N
#pragma unroll
    for (int tm = 0; tm < 4; ++tm) {
#pragma unroll
        for (int tn = 0; tn < 4; ++tn) {
            const int row0 = bm * 128 + wm * 64 + tm * 16 + h * 8;
            const int col  = bn * 256 + wn * 64 + tn * 16 + l15;
            const float bb = bias[col];
            const v8f a = acc[tm][tn];
#pragma unroll
            for (int r = 0; r < 8; ++r) {
                const size_t idx = (size_t)(row0 + r) * N + col;
                float v = a[r] + bb;
                if (EPI == 0) {
                    Cf[idx] = v;
                } else if (EPI == 1) {
                    Cf[idx] = R[idx] + v;
                } else {  // EPI == 2
                    Cb[idx] = (bf16_t)gelu_exact(v);
                }
            }
        }
    }
}

// ---------------------------------------------------------------------------
// Attention over the HEAD axis (reference quirk): per token, P = Q·K^T over
// heads (16x16, causal-over-head tril), softmax over t, attn = P·V.
// ~0.5 GFLOP total -> plain VALU in fp32. One thread per (token, head).
// ---------------------------------------------------------------------------
__global__ __launch_bounds__(256) void attention_heads(
    const float* __restrict__ Q, const float* __restrict__ K,
    const float* __restrict__ V, bf16_t* __restrict__ attn) {
    const int gid = blockIdx.x * blockDim.x + threadIdx.x;  // 0 .. M*H-1
    const int n = gid >> 4;   // token
    const int hh = gid & 15;  // head (query row)

    const float* qp = Q + (size_t)n * EE + hh * DHD;
    float q[DHD];
#pragma unroll
    for (int d = 0; d < DHD; d += 4) {
        float4 t = *(const float4*)(qp + d);
        q[d] = t.x; q[d + 1] = t.y; q[d + 2] = t.z; q[d + 3] = t.w;
    }

    float sc[HH];
    float mx = -3.4e38f;
    for (int t = 0; t <= hh; ++t) {
        const float* kp = K + (size_t)n * EE + t * DHD;
        float dot = 0.f;
#pragma unroll
        for (int d = 0; d < DHD; d += 4) {
            float4 kv = *(const float4*)(kp + d);
            dot += q[d] * kv.x + q[d + 1] * kv.y +
                   q[d + 2] * kv.z + q[d + 3] * kv.w;
        }
        sc[t] = dot * SCALE_ATTN;
        mx = fmaxf(mx, sc[t]);
    }
    float sum = 0.f;
    for (int t = 0; t <= hh; ++t) {
        sc[t] = __expf(sc[t] - mx);
        sum += sc[t];
    }
    const float inv = 1.0f / sum;

    float acc[DHD];
#pragma unroll
    for (int d = 0; d < DHD; ++d) acc[d] = 0.f;
    for (int t = 0; t <= hh; ++t) {
        const float p = sc[t] * inv;
        const float* vp = V + (size_t)n * EE + t * DHD;
#pragma unroll
        for (int d = 0; d < DHD; d += 4) {
            float4 vv = *(const float4*)(vp + d);
            acc[d]     += p * vv.x;
            acc[d + 1] += p * vv.y;
            acc[d + 2] += p * vv.z;
            acc[d + 3] += p * vv.w;
        }
    }
    bf16_t* op = attn + (size_t)n * EE + hh * DHD;
#pragma unroll
    for (int d = 0; d < DHD; ++d) op[d] = (bf16_t)acc[d];
}

// ---------------------------------------------------------------------------
// Host-side orchestration
// ---------------------------------------------------------------------------
extern "C" void kernel_launch(void* const* d_in, const int* in_sizes, int n_in,
                              void* d_out, int out_size, void* d_ws,
                              size_t ws_size, hipStream_t stream) {
    (void)in_sizes; (void)n_in; (void)out_size; (void)ws_size;

    const float* x     = (const float*)d_in[0];
    const float* ln1_w = (const float*)d_in[1];
    const float* ln1_b = (const float*)d_in[2];
    const float* Wq    = (const float*)d_in[3];
    const float* bq    = (const float*)d_in[4];
    const float* Wk    = (const float*)d_in[5];
    const float* bk    = (const float*)d_in[6];
    const float* Wv    = (const float*)d_in[7];
    const float* bv    = (const float*)d_in[8];
    const float* Wo    = (const float*)d_in[9];
    const float* bo    = (const float*)d_in[10];
    const float* ln2_w = (const float*)d_in[11];
    const float* ln2_b = (const float*)d_in[12];
    const float* W1    = (const float*)d_in[13];
    const float* b1    = (const float*)d_in[14];
    const float* W2    = (const float*)d_in[15];
    const float* b2    = (const float*)d_in[16];
    float* out = (float*)d_out;

    // Workspace carve-up (256B aligned)
    char* base = (char*)d_ws;
    size_t off = 0;
    auto carve = [&](size_t bytes) -> char* {
        char* p = base + off;
        off += (bytes + 255) & ~(size_t)255;
        return p;
    };
    bf16_t* wq_b  = (bf16_t*)carve((size_t)EE * EE * 2);
    bf16_t* wk_b  = (bf16_t*)carve((size_t)EE * EE * 2);
    bf16_t* wv_b  = (bf16_t*)carve((size_t)EE * EE * 2);
    bf16_t* wo_b  = (bf16_t*)carve((size_t)EE * EE * 2);
    bf16_t* w1_b  = (bf16_t*)carve((size_t)FF * EE * 2);
    bf16_t* w2_b  = (bf16_t*)carve((size_t)EE * FF * 2);
    bf16_t* h1_b  = (bf16_t*)carve((size_t)MM * EE * 2);
    float*  Qf    = (float*) carve((size_t)MM * EE * 4);
    float*  Kf    = (float*) carve((size_t)MM * EE * 4);
    float*  Vf    = (float*) carve((size_t)MM * EE * 4);
    bf16_t* at_b  = (bf16_t*)carve((size_t)MM * EE * 2);
    float*  x2    = (float*) carve((size_t)MM * EE * 4);
    bf16_t* h2_b  = (bf16_t*)carve((size_t)MM * EE * 2);
    bf16_t* ff1_b = (bf16_t*)carve((size_t)MM * FF * 2);

    // 1) weights -> bf16
    cvt_f32_bf16<<<2048, 256, 0, stream>>>(Wq, wq_b, EE * EE);
    cvt_f32_bf16<<<2048, 256, 0, stream>>>(Wk, wk_b, EE * EE);
    cvt_f32_bf16<<<2048, 256, 0, stream>>>(Wv, wv_b, EE * EE);
    cvt_f32_bf16<<<2048, 256, 0, stream>>>(Wo, wo_b, EE * EE);
    cvt_f32_bf16<<<4096, 256, 0, stream>>>(W1, w1_b, FF * EE);
    cvt_f32_bf16<<<4096, 256, 0, stream>>>(W2, w2_b, EE * FF);

    // 2) LN1 -> bf16
    layernorm_bf16<<<MM, 256, 0, stream>>>(x, ln1_w, ln1_b, h1_b);

    // 3) Q/K/V projections (WMMA), fp32 out
    {
        dim3 grid(MM / 128, EE / 256);
        gemm_bf16_wmma<0><<<grid, 256, 0, stream>>>(h1_b, wq_b, bq, nullptr,
                                                    Qf, nullptr, EE, EE);
        gemm_bf16_wmma<0><<<grid, 256, 0, stream>>>(h1_b, wk_b, bk, nullptr,
                                                    Kf, nullptr, EE, EE);
        gemm_bf16_wmma<0><<<grid, 256, 0, stream>>>(h1_b, wv_b, bv, nullptr,
                                                    Vf, nullptr, EE, EE);
    }

    // 4) attention over head axis -> bf16
    attention_heads<<<(MM * HH) / 256, 256, 0, stream>>>(Qf, Kf, Vf, at_b);

    // 5) x2 = x + attn @ Wo^T + bo
    {
        dim3 grid(MM / 128, EE / 256);
        gemm_bf16_wmma<1><<<grid, 256, 0, stream>>>(at_b, wo_b, bo, x,
                                                    x2, nullptr, EE, EE);
    }

    // 6) LN2 -> bf16
    layernorm_bf16<<<MM, 256, 0, stream>>>(x2, ln2_w, ln2_b, h2_b);

    // 7) ffn1 = gelu(h2 @ W1^T + b1) -> bf16
    {
        dim3 grid(MM / 128, FF / 256);
        gemm_bf16_wmma<2><<<grid, 256, 0, stream>>>(h2_b, w1_b, b1, nullptr,
                                                    nullptr, ff1_b, FF, EE);
    }

    // 8) out = x2 + ffn1 @ W2^T + b2
    {
        dim3 grid(MM / 128, EE / 256);
        gemm_bf16_wmma<1><<<grid, 256, 0, stream>>>(ff1_b, w2_b, b2, x2,
                                                    out, nullptr, EE, FF);
    }
}
